// ChoreographModel_28363964023652
// MI455X (gfx1250) — matrix operand order repaired and checked by reading, not compiled, not run on hardware
//
#include <hip/hip_runtime.h>
#include <math.h>

typedef __attribute__((ext_vector_type(16))) _Float16 v16h;
typedef __attribute__((ext_vector_type(8)))  _Float16 h8;
typedef __attribute__((ext_vector_type(8)))  float    v8f;
typedef __attribute__((ext_vector_type(4)))  int      v4i;

#define BATCH 4096
#define HID   512
#define NCLS  256
#define TSTEPS 7
#define FEAT  160

// ---------- CDNA5 async global->LDS support (probe via __has_builtin) ----------
#if defined(__has_builtin)
#if __has_builtin(__builtin_amdgcn_global_load_async_to_lds_b128)
#define ASYNC_LDS 1
#endif
#endif
#ifndef ASYNC_LDS
#define ASYNC_LDS 0
#endif

#if ASYNC_LDS
typedef __attribute__((address_space(1))) v4i GV4;   // global v4i
typedef __attribute__((address_space(3))) v4i LV4;   // LDS v4i
__device__ __forceinline__ void async_copy16(const void* g, void* l) {
    // per-lane 16B copy, tracked by ASYNCcnt; completes in order
    __builtin_amdgcn_global_load_async_to_lds_b128((GV4*)g, (LV4*)l, 0, 0);
}
template <int N>
__device__ __forceinline__ void wait_async() {
#if defined(__has_builtin)
#if __has_builtin(__builtin_amdgcn_s_wait_asynccnt)
    __builtin_amdgcn_s_wait_asynccnt(N);
#else
    asm volatile("s_wait_asynccnt %0" ::"n"(N) : "memory");
#endif
#else
    asm volatile("s_wait_asynccnt %0" ::"n"(N) : "memory");
#endif
}
#endif

// ---------------- f32 -> f16 weight conversion ----------------
__global__ void cvt_f32_f16(const float* __restrict__ src, _Float16* __restrict__ dst, int n) {
    int i = blockIdx.x * 256 + threadIdx.x;
    if (i < n) dst[i] = (_Float16)src[i];
}

// ---------------- conv1 (7x3, 3->10) + relu + maxpool(1,3) ----------------
// x: [4096,15,80,3]  out: [4096,9,26,10]
__global__ void conv1_pool(const float* __restrict__ x, const float* __restrict__ w,
                           const float* __restrict__ bias, float* __restrict__ out) {
    int idx = blockIdx.x * 256 + threadIdx.x;              // 4096*9*26*10 exact
    int oc = idx % 10;
    int w0 = (idx / 10) % 26;
    int h  = (idx / 260) % 9;
    int b  = idx / 2340;
    float m = -3.4e38f;
    for (int pw = 0; pw < 3; ++pw) {
        int ww = w0 * 3 + pw;
        float acc = bias[oc];
        #pragma unroll
        for (int kh = 0; kh < 7; ++kh)
            #pragma unroll
            for (int kw = 0; kw < 3; ++kw)
                #pragma unroll
                for (int ic = 0; ic < 3; ++ic)
                    acc += x[((b * 15 + h + kh) * 80 + ww + kw) * 3 + ic] *
                           w[((kh * 3 + kw) * 3 + ic) * 10 + oc];
        m = fmaxf(m, fmaxf(acc, 0.f));
    }
    out[idx] = m;
}

// ---------------- conv2 (3x3, 10->20) + relu + maxpool(1,3), write seq f16 ----------------
// in: [4096,9,26,10]  seq: [7][4096][160] f16
__global__ void conv2_pool(const float* __restrict__ in, const float* __restrict__ w,
                           const float* __restrict__ bias, _Float16* __restrict__ seq) {
    int idx = blockIdx.x * 256 + threadIdx.x;              // 4096*7*8*20 exact
    int oc = idx % 20;
    int w0 = (idx / 20) % 8;
    int h  = (idx / 160) % 7;
    int b  = idx / 1120;
    float m = -3.4e38f;
    for (int pw = 0; pw < 3; ++pw) {
        int ww = w0 * 3 + pw;
        float acc = bias[oc];
        #pragma unroll
        for (int kh = 0; kh < 3; ++kh)
            #pragma unroll
            for (int kw = 0; kw < 3; ++kw)
                #pragma unroll
                for (int ic = 0; ic < 10; ++ic)
                    acc += in[((b * 9 + h + kh) * 26 + ww + kw) * 10 + ic] *
                           w[((kh * 3 + kw) * 10 + ic) * 20 + oc];
        m = fmaxf(m, fmaxf(acc, 0.f));
    }
    seq[((size_t)h * BATCH + b) * FEAT + w0 * 20 + oc] = (_Float16)m;
}

// ---------------- dual-input WMMA GEMM: C = A0@B0 + A1@B1 + bias (+relu) ----------------
// Double-buffered LDS software pipeline; A tile via async global->LDS when available.
#define BM 128
#define BN 128
#define BK 32
#define LDT 40   // padded LDS stride (halfs)

__global__ __launch_bounds__(256) void gemm_dual_wmma(
    const _Float16* __restrict__ A0, int K0, const _Float16* __restrict__ B0,
    const _Float16* __restrict__ A1, int K1, const _Float16* __restrict__ B1,
    const float* __restrict__ bias, float* __restrict__ C, int N, int relu) {
    __shared__ _Float16 sA[2][BM][LDT];   // [buf][m][k]
    __shared__ _Float16 sB[2][BN][LDT];   // [buf][n][k]  (transposed)

    const int tid  = threadIdx.x;
    const int lane = tid & 31;
    const int wv   = tid >> 5;
    const int wm   = (wv >> 2) * 64;      // 0 | 64
    const int wn   = (wv & 3) * 32;       // 0,32,64,96
    const int m0   = blockIdx.y * BM;
    const int n0   = blockIdx.x * BN;

    const int colA = (tid & 3) * 8;       // A tile: 128x32, 2x h8 per thread
    const int rA   = tid >> 2;
    const int nB   = (tid & 15) * 8;      // B tile: 32x128, rows kB and kB+16
    const int kB   = tid >> 4;

    v8f acc[4][2];
    #pragma unroll
    for (int i = 0; i < 4; ++i)
        #pragma unroll
        for (int j = 0; j < 2; ++j) {
            v8f z = {0.f, 0.f, 0.f, 0.f, 0.f, 0.f, 0.f, 0.f};
            acc[i][j] = z;
        }

    const int t0 = K0 / BK, t1 = K1 / BK, T = t0 + t1;

    h8 rB0, rB1;            // B stage registers (next tile)
#if !ASYNC_LDS
    h8 rA0, rA1;            // A stage registers (sync fallback)
#endif

    auto issue = [&](int kt, int bufsel) {
        const _Float16* Ag;
        const _Float16* Bg;
        int lda, kk;
        if (kt < t0) { Ag = A0; Bg = B0; lda = K0; kk = kt * BK; }
        else         { Ag = A1; Bg = B1; lda = K1; kk = (kt - t0) * BK; }
        const _Float16* g0 = Ag + (size_t)(m0 + rA) * lda + kk + colA;
        const _Float16* g1 = Ag + (size_t)(m0 + rA + 64) * lda + kk + colA;
#if ASYNC_LDS
        async_copy16(g0, &sA[bufsel][rA][colA]);
        async_copy16(g1, &sA[bufsel][rA + 64][colA]);
#else
        (void)bufsel;
        rA0 = *(const h8*)g0;
        rA1 = *(const h8*)g1;
#endif
        rB0 = *(const h8*)(Bg + (size_t)(kk + kB) * N + n0 + nB);
        rB1 = *(const h8*)(Bg + (size_t)(kk + kB + 16) * N + n0 + nB);
    };

    issue(0, 0);   // preload stage 0

    for (int kt = 0; kt < T; ++kt) {
        const int b = kt & 1;

        // ---- stage-kt staged data -> LDS buffer b ----
        h8 b0 = rB0, b1 = rB1;
#if !ASYNC_LDS
        h8 a0 = rA0, a1 = rA1;
        *(h8*)&sA[b][rA][colA]      = a0;
        *(h8*)&sA[b][rA + 64][colA] = a1;
#endif
        #pragma unroll
        for (int j = 0; j < 8; ++j) {
            sB[b][nB + j][kB]      = b0[j];
            sB[b][nB + j][kB + 16] = b1[j];
        }

        // ---- issue stage kt+1 into the other buffer ----
        if (kt + 1 < T) issue(kt + 1, b ^ 1);

#if ASYNC_LDS
        // async copies complete in order: after issuing 2 more, <=2 outstanding
        // means stage kt's two copies have landed.
        if (kt + 1 < T) wait_async<2>();
        else            wait_async<0>();
#endif
        __syncthreads();

        // ---- fragments per ISA layout (lane&15 = row/col, lane>>4 = K half) ----
        v16h af[4], bf[2];
        const int r   = lane & 15;
        const int kb  = (lane >> 4) * 8;    // A: K 0..7 / 8..15 (+16 for hi half)

        #pragma unroll
        for (int tm = 0; tm < 4; ++tm) {
            const _Float16* p0 = &sA[b][wm + tm * 16 + r][kb];
            #pragma unroll
            for (int j = 0; j < 8; ++j) { af[tm][j] = p0[j]; af[tm][8 + j] = p0[16 + j]; }
        }
        #pragma unroll
        for (int tn = 0; tn < 2; ++tn) {
            const _Float16* p = &sB[b][wn + tn * 16 + r][kb];
            #pragma unroll
            for (int j = 0; j < 8; ++j) { bf[tn][j] = p[j]; bf[tn][8 + j] = p[16 + j]; }
        }
        #pragma unroll
        for (int tm = 0; tm < 4; ++tm)
            #pragma unroll
            for (int tn = 0; tn < 2; ++tn)
                acc[tm][tn] = __builtin_amdgcn_wmma_f32_16x16x32_f16(
                    false, af[tm], false, bf[tn], (short)0, acc[tm][tn], false, false);
        __syncthreads();
    }

    // ---- epilogue: bias (+relu); D layout: n=lane&15, M = r + 8*(lane>>4) ----
    const int col  = lane & 15;
    const int rsel = (lane >> 4) * 8;
    #pragma unroll
    for (int tm = 0; tm < 4; ++tm)
        #pragma unroll
        for (int tn = 0; tn < 2; ++tn) {
            int   ng = n0 + wn + tn * 16 + col;
            float bv = bias[ng];
            #pragma unroll
            for (int rr = 0; rr < 8; ++rr) {
                int   mg = m0 + wm + tm * 16 + rsel + rr;
                float v  = acc[tm][tn][rr] + bv;
                if (relu) v = fmaxf(v, 0.f);
                C[(size_t)mg * N + ng] = v;
            }
        }
}

// ---------------- LSTM pointwise cell: z[B,4H] (gate order i,f,g,o) ----------------
__global__ void lstm_cell(const float* __restrict__ z, float* __restrict__ c,
                          _Float16* __restrict__ h) {
    int idx = blockIdx.x * 256 + threadIdx.x;   // 4096*512 exact
    int b = idx >> 9;
    int j = idx & 511;
    const float* zr = z + (size_t)b * (4 * HID);
    float zi = zr[j], zf = zr[HID + j], zg = zr[2 * HID + j], zo = zr[3 * HID + j];
    float si = 1.f / (1.f + __expf(-zi));
    float sf = 1.f / (1.f + __expf(-zf));
    float so = 1.f / (1.f + __expf(-zo));
    float cn = sf * c[idx] + si * tanhf(zg);
    c[idx] = cn;
    h[idx] = (_Float16)(so * tanhf(cn));
}

// ---------------- launcher ----------------
extern "C" void kernel_launch(void* const* d_in, const int* in_sizes, int n_in,
                              void* d_out, int out_size, void* d_ws, size_t ws_size,
                              hipStream_t stream) {
    const float* x       = (const float*)d_in[0];
    const float* conv1_w = (const float*)d_in[1];
    const float* conv1_b = (const float*)d_in[2];
    const float* conv2_w = (const float*)d_in[3];
    const float* conv2_b = (const float*)d_in[4];
    const float* W1      = (const float*)d_in[5];
    const float* U1      = (const float*)d_in[6];
    const float* b1      = (const float*)d_in[7];
    const float* W2      = (const float*)d_in[8];
    const float* U2      = (const float*)d_in[9];
    const float* b2      = (const float*)d_in[10];
    const float* fc_w    = (const float*)d_in[11];
    const float* fc_b    = (const float*)d_in[12];

    char*  ws  = (char*)d_ws;
    size_t off = 0;
    auto alloc = [&](size_t bytes) -> void* {
        void* p = ws + off;
        off = (off + bytes + 255) & ~(size_t)255;
        return p;
    };

    _Float16* seqh  = (_Float16*)alloc((size_t)TSTEPS * BATCH * FEAT * 2);
    float*    pool1 = (float*)   alloc((size_t)BATCH * 9 * 26 * 10 * 4);
    _Float16* W1h   = (_Float16*)alloc((size_t)FEAT * 4 * HID * 2);
    _Float16* U1h   = (_Float16*)alloc((size_t)HID * 4 * HID * 2);
    _Float16* W2h   = (_Float16*)alloc((size_t)HID * 4 * HID * 2);
    _Float16* U2h   = (_Float16*)alloc((size_t)HID * 4 * HID * 2);
    _Float16* FCh   = (_Float16*)alloc((size_t)HID * NCLS * 2);
    _Float16* h1    = (_Float16*)alloc((size_t)BATCH * HID * 2);
    _Float16* h2    = (_Float16*)alloc((size_t)BATCH * HID * 2);
    float*    c1    = (float*)   alloc((size_t)BATCH * HID * 4);
    float*    c2    = (float*)   alloc((size_t)BATCH * HID * 4);
    float*    zbuf  = (float*)   alloc((size_t)BATCH * 4 * HID * 4);

    // weight conversions (f32 -> f16)
    cvt_f32_f16<<<(FEAT * 4 * HID + 255) / 256, 256, 0, stream>>>(W1, W1h, FEAT * 4 * HID);
    cvt_f32_f16<<<(HID * 4 * HID + 255) / 256, 256, 0, stream>>>(U1, U1h, HID * 4 * HID);
    cvt_f32_f16<<<(HID * 4 * HID + 255) / 256, 256, 0, stream>>>(W2, W2h, HID * 4 * HID);
    cvt_f32_f16<<<(HID * 4 * HID + 255) / 256, 256, 0, stream>>>(U2, U2h, HID * 4 * HID);
    cvt_f32_f16<<<(HID * NCLS + 255) / 256, 256, 0, stream>>>(fc_w, FCh, HID * NCLS);

    // conv stack
    conv1_pool<<<(BATCH * 9 * 26 * 10) / 256, 256, 0, stream>>>(x, conv1_w, conv1_b, pool1);
    conv2_pool<<<(BATCH * 7 * 8 * 20) / 256, 256, 0, stream>>>(pool1, conv2_w, conv2_b, seqh);

    // zero recurrent state every call (ws is poisoned)
    (void)hipMemsetAsync(h1, 0, (size_t)BATCH * HID * 2, stream);
    (void)hipMemsetAsync(h2, 0, (size_t)BATCH * HID * 2, stream);
    (void)hipMemsetAsync(c1, 0, (size_t)BATCH * HID * 4, stream);
    (void)hipMemsetAsync(c2, 0, (size_t)BATCH * HID * 4, stream);

    const dim3 blk(256);
    const dim3 grid_z(4 * HID / BN, BATCH / BM);   // (16, 32)
    const dim3 grid_fc(NCLS / BN, BATCH / BM);     // (2, 32)
    const int  cell_blocks = (BATCH * HID) / 256;

    for (int t = 0; t < TSTEPS; ++t) {
        const _Float16* xt = seqh + (size_t)t * BATCH * FEAT;
        // z1 = x_t@W1 + h1@U1 + b1
        gemm_dual_wmma<<<grid_z, blk, 0, stream>>>(xt, FEAT, W1h, h1, HID, U1h,
                                                   b1, zbuf, 4 * HID, 0);
        lstm_cell<<<cell_blocks, blk, 0, stream>>>(zbuf, c1, h1);
        // z2 = h1@W2 + h2@U2 + b2
        gemm_dual_wmma<<<grid_z, blk, 0, stream>>>(h1, HID, W2h, h2, HID, U2h,
                                                   b2, zbuf, 4 * HID, 0);
        lstm_cell<<<cell_blocks, blk, 0, stream>>>(zbuf, c2, h2);
    }

    // out = relu(h2 @ fc_w + fc_b)  (A1 path disabled: K1 = 0)
    gemm_dual_wmma<<<grid_fc, blk, 0, stream>>>(h2, HID, FCh, h2, 0, FCh,
                                                fc_b, (float*)d_out, NCLS, 1);
}
